// Fourier_15968688406806
// MI455X (gfx1250) — matrix operand order repaired and checked by reading, not compiled, not run on hardware
//
#include <hip/hip_runtime.h>
#include <hip/hip_bf16.h>

typedef __attribute__((ext_vector_type(16))) _Float16 v16h;
typedef __attribute__((ext_vector_type(8)))  float    v8f;

// Shapes from the reference
#define BB 262144
#define II 3
#define OO 64
#define GG 128
#define KTOT 768          // 2 * I * G, K-interleaved: even k = cos, odd k = sin
#define KSTEPS 24         // 768 / 32

// 256 threads = 8 waves; each wave: 16-row M tile x all four 16-col N tiles.
#define ROWS_PER_BLOCK 128

#define INV_2PI 0.15915494309189535f

#if defined(__has_builtin)
#  if __has_builtin(__builtin_amdgcn_sinf) && __has_builtin(__builtin_amdgcn_cosf)
#    define HW_TRIG 1
#  endif
#endif

// Compute (cos(2*pi*rev), sin(2*pi*rev)) from a pre-scaled "revolutions" arg.
__device__ __forceinline__ void sincos_rev(float rev, float* s, float* c) {
#ifdef HW_TRIG
    *c = __builtin_amdgcn_cosf(rev);   // v_cos_f32: cos(2*pi*rev)
    *s = __builtin_amdgcn_sinf(rev);   // v_sin_f32: sin(2*pi*rev)
#else
    __sincosf(rev * 6.283185307179586f, s, c);
#endif
}

__global__ __launch_bounds__(256)
void fourier_wmma_kernel(const float* __restrict__ x,
                         const float* __restrict__ coeffs,
                         const float* __restrict__ bias,
                         float* __restrict__ out) {
    // ---- Stage all weights into LDS as f16 ----
    // K mapping   : k = 2*j + br, j = i*128 + g, br: 0=cos weights, 1=sin weights
    //               -> packed f16 pair (k, k+1) multiplies (cos θ_j, sin θ_j).
    // N permutation: WMMA tile nt, tile-local col c  <-> global col n = 4*c + nt.
    // LDS layout  : ldsW[(k>>1)*128 + 2*nperm + (k&1)], nperm = (n&3)*16 + (n>>2)
    __shared__ _Float16 ldsW[KTOT * OO];   // 49152 halves = 96 KB

    const int tid = threadIdx.x;

    for (int t = tid; t < KTOT * OO; t += 256) {
        const int n  = t / KTOT;
        const int k  = t - n * KTOT;         // consecutive lanes -> consecutive k
        const int br = k & 1;                // 0 = cos branch, 1 = sin branch
        const int j  = k >> 1;               // 0..383
        const int ii = j >> 7;               // input dim 0..2
        const int g  = j & 127;              // grid index
        const float w = coeffs[(((br * OO + n) * II) + ii) * GG + g];
        const int nperm = (n & 3) * 16 + (n >> 2);
        ldsW[((k >> 1) << 7) + (nperm << 1) + br] = (_Float16)w;
    }
    __syncthreads();

    const int lane  = tid & 31;
    const int wid   = tid >> 5;        // wave 0..7
    const int half  = lane >> 4;       // 0 or 1
    const int nlane = lane & 15;       // tile-local N / A-tile M
    const int half4 = half << 2;
    const int half8 = half << 3;

    const int rowBase = blockIdx.x * ROWS_PER_BLOCK + wid * 16;
    const int myRow   = rowBase + nlane;

    const float x0 = x[myRow * 3 + 0];
    const float x1 = x[myRow * 3 + 1];
    const float x2 = x[myRow * 3 + 2];

    v8f acc[4];
    #pragma unroll
    for (int nt = 0; nt < 4; ++nt) acc[nt] = (v8f){0.f,0.f,0.f,0.f,0.f,0.f,0.f,0.f};

    #pragma unroll
    for (int kk = 0; kk < KSTEPS; ++kk) {
        const int jb = kk * 16;                // angle-index window base (uniform)
        const int ii = jb >> 7;                // which coordinate, uniform per kk
        const int gb = jb & 127;               // grid base, uniform per kk
        const float xi = (ii == 0) ? x0 : ((ii == 1) ? x1 : x2);

        // Pre-scaled "revolutions" form: angle = (gb + dj + 1) * xi
        //  -> rev = (gb + dj + 1) * xi / (2*pi) = fma(dj, xs, base)
        const float xs   = xi * INV_2PI;               // 1 mul per window
        const float base = (float)(gb + 1) * xs;       // 1 mul per window

        // ---- A fragment: VGPR v holds packed (cos θ, sin θ) of one angle ----
        v16h a;
        #pragma unroll
        for (int v = 0; v < 8; ++v) {
            const int dj = v + half8;
            const float rev = __builtin_fmaf((float)dj, xs, base);
            float s, c;
            sincos_rev(rev, &s, &c);
            a[2 * v]     = (_Float16)c;   // even k -> cos
            a[2 * v + 1] = (_Float16)s;   // odd  k -> sin
        }

        // ---- 4 N-tiles: B fragment from LDS, WMMA ----
        #pragma unroll
        for (int nt = 0; nt < 4; ++nt) {
            const int nG   = nt * 16 + nlane;          // permuted column slot
            const int base2 = (kk * 16 + half * 8) * 128 + 2 * nG;
            v16h bf;
            #pragma unroll
            for (int v = 0; v < 8; ++v) {
                bf[2 * v]     = ldsW[base2 + v * 128];
                bf[2 * v + 1] = ldsW[base2 + v * 128 + 1];
            }
            acc[nt] = __builtin_amdgcn_wmma_f32_16x16x32_f16(
                /*neg_a=*/false, a, /*neg_b=*/false, bf,
                /*c_mod=*/(short)0, acc[nt],
                /*reuse_a=*/false, /*reuse_b=*/false);
        }
    }

    // ---- Epilogue: lane holds cols 4*nlane..4*nlane+3 -> float4 stores ----
    const float4 bv = ((const float4*)bias)[nlane];
    float4* out4 = (float4*)out;
    #pragma unroll
    for (int r = 0; r < 8; ++r) {
        const int row = rowBase + half8 + r;
        float4 o;
        o.x = acc[0][r] + bv.x;
        o.y = acc[1][r] + bv.y;
        o.z = acc[2][r] + bv.z;
        o.w = acc[3][r] + bv.w;
        out4[(size_t)row * (OO / 4) + nlane] = o;
    }
}

extern "C" void kernel_launch(void* const* d_in, const int* in_sizes, int n_in,
                              void* d_out, int out_size, void* d_ws, size_t ws_size,
                              hipStream_t stream) {
    const float* x      = (const float*)d_in[0];   // (1, B, 3)
    const float* coeffs = (const float*)d_in[1];   // (2, 64, 3, 128)
    const float* bias   = (const float*)d_in[2];   // (1, 64)
    float* out          = (float*)d_out;           // (B, 64)

    const int grid = BB / ROWS_PER_BLOCK;          // 2048 blocks
    fourier_wmma_kernel<<<grid, 256, 0, stream>>>(x, coeffs, bias, out);
}
